// Qwen2MoeEagerSparseMoeBlock_12378095747249
// MI455X (gfx1250) — matrix operand order repaired and checked by reading, not compiled, not run on hardware
//
#include <hip/hip_runtime.h>
#include <hip/hip_bf16.h>
#include <stdint.h>

// ---------------------------------------------------------------------------
// Qwen2 MoE sparse block for gfx1250 (MI455X): bf16 WMMA GEMM pipeline.
// ---------------------------------------------------------------------------

#define H_DIM   2048
#define I_MOE   1408
#define I_SH    5632
#define N_EXP   8
#define T_TOK   2048

typedef __attribute__((ext_vector_type(16))) __bf16 v16bf;
typedef __attribute__((ext_vector_type(8)))  __bf16 v8bf;
typedef __attribute__((ext_vector_type(4)))  __bf16 v4bf;
typedef __attribute__((ext_vector_type(8)))  float  v8f;
typedef __attribute__((ext_vector_type(4)))  float  v4f;

union BFrag { v16bf v; v8bf h[2]; };

__device__ __forceinline__ __bf16 f2bf(float f) {
    // fp32 -> bf16 round-to-nearest-even
    union { float f; uint32_t u; } v; v.f = f;
    uint32_t r = v.u + 0x7FFFu + ((v.u >> 16) & 1u);
    union { uint16_t s; __bf16 b; } o; o.s = (uint16_t)(r >> 16);
    return o.b;
}

__device__ __forceinline__ v4bf f2bf4(v4f f) {
    v4bf o;
    o.x = f2bf(f.x); o.y = f2bf(f.y); o.z = f2bf(f.z); o.w = f2bf(f.w);
    return o;
}

// GEMM tiling: 128x128 block tile, K-step 32, 8 waves (wave32) per block.
// Wave grid 4(M) x 2(N): each wave computes 32(M) x 64(N) via 2x4 WMMA tiles.
#define BM 128
#define BN 128
#define BK 32
#define LDSS 40   // padded row stride (bf16): 80B = 5*16B -> aligned b128 frags

// ---------------------------------------------------------------------------
// Kernel 1: router softmax/top-2 (-> dense combine matrix) + shared-gate sigmoid
// ---------------------------------------------------------------------------
__global__ __launch_bounds__(256)
void router_kernel(const float* __restrict__ X, const float* __restrict__ Wr,
                   const float* __restrict__ Wsg,
                   float* __restrict__ combine, float* __restrict__ sgate_sig)
{
    const int t   = blockIdx.x;
    const int tid = threadIdx.x;

    float part[9];
#pragma unroll
    for (int j = 0; j < 9; ++j) part[j] = 0.0f;

    const float* xr = X + (size_t)t * H_DIM;
    for (int h = tid; h < H_DIM; h += 256) {
        float xv = xr[h];
#pragma unroll
        for (int e = 0; e < N_EXP; ++e) part[e] += xv * Wr[(size_t)e * H_DIM + h];
        part[8] += xv * Wsg[h];
    }

    __shared__ float red[9][256];
#pragma unroll
    for (int j = 0; j < 9; ++j) red[j][tid] = part[j];
    __syncthreads();
    for (int s = 128; s > 0; s >>= 1) {
        if (tid < s) {
#pragma unroll
            for (int j = 0; j < 9; ++j) red[j][tid] += red[j][tid + s];
        }
        __syncthreads();
    }

    if (tid == 0) {
        float p[N_EXP];
        float m = -1e30f;
#pragma unroll
        for (int e = 0; e < N_EXP; ++e) { p[e] = red[e][0]; m = fmaxf(m, p[e]); }
        float sum = 0.0f;
#pragma unroll
        for (int e = 0; e < N_EXP; ++e) { p[e] = __expf(p[e] - m); sum += p[e]; }
        float inv = 1.0f / sum;
#pragma unroll
        for (int e = 0; e < N_EXP; ++e) p[e] *= inv;

        int i1 = 0;
#pragma unroll
        for (int e = 1; e < N_EXP; ++e) if (p[e] > p[i1]) i1 = e;
        int i2 = (i1 == 0) ? 1 : 0;
#pragma unroll
        for (int e = 0; e < N_EXP; ++e) if (e != i1 && p[e] > p[i2]) i2 = e;

        float* crow = combine + (size_t)t * N_EXP;
#pragma unroll
        for (int e = 0; e < N_EXP; ++e)
            crow[e] = (e == i1) ? p[i1] : ((e == i2) ? p[i2] : 0.0f);

        sgate_sig[t] = 1.0f / (1.0f + __expf(-red[8][0]));
    }
}

// ---------------------------------------------------------------------------
// Staging helpers (vectorized: float4 global loads, packed bf16 LDS stores)
// ---------------------------------------------------------------------------
// A tile: [BM x BK] fp32 -> bf16 row-major (stride LDSS). 4x float4 per thread.
__device__ __forceinline__ void stage_A(__bf16* __restrict__ As,
                                        const float* __restrict__ X,
                                        int m0, int K, int k0, int tid)
{
#pragma unroll
    for (int p = 0; p < 4; ++p) {
        int idx4 = p * 256 + tid;          // float4 index within 128x8 grid
        int r = idx4 >> 3;                 // row 0..127
        int c = (idx4 & 7) << 2;           // col 0,4,..,28
        v4f v = *(const v4f*)(X + (size_t)(m0 + r) * K + k0 + c);
        *(v4bf*)(&As[r * LDSS + c]) = f2bf4(v);
    }
}

// B tile: [BK x BN] fp32 -> bf16 transposed Bs[n][k]. 4x float4 per thread.
__device__ __forceinline__ void stage_Bt(__bf16* __restrict__ Bs,
                                         const float* __restrict__ W,
                                         int k0, int ldb, int n0, int tid)
{
#pragma unroll
    for (int p = 0; p < 4; ++p) {
        int idx4 = p * 256 + tid;          // float4 index within 32x32 grid
        int k = idx4 >> 5;                 // k 0..31
        int n = (idx4 & 31) << 2;          // n 0,4,..,124
        v4f v = *(const v4f*)(W + (size_t)(k0 + k) * ldb + n0 + n);
        Bs[(n + 0) * LDSS + k] = f2bf(v.x);
        Bs[(n + 1) * LDSS + k] = f2bf(v.y);
        Bs[(n + 2) * LDSS + k] = f2bf(v.z);
        Bs[(n + 3) * LDSS + k] = f2bf(v.w);
    }
}

// Load one 16x32 bf16 A fragment (documented 16-bit A layout).
__device__ __forceinline__ void load_afrag(BFrag& a, const __bf16* __restrict__ As,
                                           int rowBase, int laneLo, int hi)
{
    const __bf16* pa = &As[(rowBase + laneLo) * LDSS + hi * 8];
    a.h[0] = *(const v8bf*)(pa);
    a.h[1] = *(const v8bf*)(pa + 16);
}

// Load one 32x16 bf16 B fragment (documented 16-bit B layout).
__device__ __forceinline__ void load_bfrag(BFrag& b, const __bf16* __restrict__ Bs,
                                           int colBase, int laneLo, int hi)
{
    const __bf16* pb = &Bs[(colBase + laneLo) * LDSS + hi * 8];
    b.h[0] = *(const v8bf*)(pb);
    b.h[1] = *(const v8bf*)(pb + 16);
}

// ---------------------------------------------------------------------------
// Kernel 2: fused gate/up GEMM + SiLU*up (+ optional per-row combine scale).
//   X:[M x K] row-major, W:[K x ldb] (gate cols [0,Itot), up cols [Itot,2*Itot))
//   Hout:[M x Itot].  scale!=nullptr -> Hout row r *= scale[r*scaleStride].
// ---------------------------------------------------------------------------
__global__ __launch_bounds__(256)
void gemm_gateup_silu(const float* __restrict__ X, const float* __restrict__ W,
                      float* __restrict__ Hout, const float* __restrict__ scale,
                      int K, int Itot, int ldb, int scaleStride)
{
    __shared__ __bf16 As [BM * LDSS];
    __shared__ __bf16 BsG[BN * LDSS];
    __shared__ __bf16 BsU[BN * LDSS];

    const int tid    = threadIdx.x;
    const int lane   = tid & 31;
    const int wave   = tid >> 5;
    const int wm     = wave >> 1;      // 0..3
    const int wn     = wave & 1;       // 0..1
    const int laneLo = lane & 15;
    const int hi     = lane >> 4;      // half-wave select
    const int m0     = blockIdx.y * BM;
    const int n0     = blockIdx.x * BN;

    v8f accG[2][4], accU[2][4];
    const v8f vzero = {0.f, 0.f, 0.f, 0.f, 0.f, 0.f, 0.f, 0.f};
#pragma unroll
    for (int mi = 0; mi < 2; ++mi)
#pragma unroll
        for (int ni = 0; ni < 4; ++ni) { accG[mi][ni] = vzero; accU[mi][ni] = vzero; }

    for (int k0 = 0; k0 < K; k0 += BK) {
        __syncthreads();
        if (k0 + BK < K) {  // hint next K-tile into cache (global_prefetch_b8)
            __builtin_prefetch(X + (size_t)(m0 + (tid >> 1)) * K + k0 + BK, 0, 1);
            __builtin_prefetch(W + (size_t)(k0 + BK + (tid >> 4)) * ldb + n0, 0, 1);
        }
        stage_A (As,  X, m0, K, k0, tid);
        stage_Bt(BsG, W,           k0, ldb, n0, tid);
        stage_Bt(BsU, W + Itot,    k0, ldb, n0, tid);
        __syncthreads();

        BFrag a[2];
#pragma unroll
        for (int mi = 0; mi < 2; ++mi)
            load_afrag(a[mi], As, wm * 32 + mi * 16, laneLo, hi);
#pragma unroll
        for (int ni = 0; ni < 4; ++ni) {
            BFrag bg, bu;
            load_bfrag(bg, BsG, wn * 64 + ni * 16, laneLo, hi);
            load_bfrag(bu, BsU, wn * 64 + ni * 16, laneLo, hi);
#pragma unroll
            for (int mi = 0; mi < 2; ++mi) {
                accG[mi][ni] = __builtin_amdgcn_wmma_f32_16x16x32_bf16(
                    false, a[mi].v, false, bg.v, (short)0, accG[mi][ni], false, false);
                accU[mi][ni] = __builtin_amdgcn_wmma_f32_16x16x32_bf16(
                    false, a[mi].v, false, bu.v, (short)0, accU[mi][ni], false, false);
            }
        }
    }

    // Epilogue: h = silu(g) * u [* combine]
#pragma unroll
    for (int mi = 0; mi < 2; ++mi)
#pragma unroll
        for (int ni = 0; ni < 4; ++ni) {
            int col = n0 + wn * 64 + ni * 16 + laneLo;
#pragma unroll
            for (int r = 0; r < 8; ++r) {
                int row = m0 + wm * 32 + mi * 16 + hi * 8 + r;
                float g = accG[mi][ni][r];
                float u = accU[mi][ni][r];
                float hv = (g / (1.0f + __expf(-g))) * u;
                if (scale) hv *= scale[(size_t)row * scaleStride];
                Hout[(size_t)row * Itot + col] = hv;
            }
        }
}

// ---------------------------------------------------------------------------
// Kernel 3: down-projection GEMM. out = A @ W, optionally *rowScale[row],
// optionally accumulated into out (sequential launches -> race-free).
// ---------------------------------------------------------------------------
__global__ __launch_bounds__(256)
void gemm_down(const float* __restrict__ A, const float* __restrict__ W,
               float* __restrict__ out, const float* __restrict__ rowScale,
               int K, int N, int accumulate)
{
    __shared__ __bf16 As[BM * LDSS];
    __shared__ __bf16 Bs[BN * LDSS];

    const int tid    = threadIdx.x;
    const int lane   = tid & 31;
    const int wave   = tid >> 5;
    const int wm     = wave >> 1;
    const int wn     = wave & 1;
    const int laneLo = lane & 15;
    const int hi     = lane >> 4;
    const int m0     = blockIdx.y * BM;
    const int n0     = blockIdx.x * BN;

    v8f acc[2][4];
    const v8f vzero = {0.f, 0.f, 0.f, 0.f, 0.f, 0.f, 0.f, 0.f};
#pragma unroll
    for (int mi = 0; mi < 2; ++mi)
#pragma unroll
        for (int ni = 0; ni < 4; ++ni) acc[mi][ni] = vzero;

    for (int k0 = 0; k0 < K; k0 += BK) {
        __syncthreads();
        if (k0 + BK < K) {
            __builtin_prefetch(A + (size_t)(m0 + (tid >> 1)) * K + k0 + BK, 0, 1);
            __builtin_prefetch(W + (size_t)(k0 + BK + (tid >> 4)) * N + n0, 0, 1);
        }
        stage_A (As, A, m0, K, k0, tid);
        stage_Bt(Bs, W,          k0, N, n0, tid);
        __syncthreads();

        BFrag a[2];
#pragma unroll
        for (int mi = 0; mi < 2; ++mi)
            load_afrag(a[mi], As, wm * 32 + mi * 16, laneLo, hi);
#pragma unroll
        for (int ni = 0; ni < 4; ++ni) {
            BFrag b;
            load_bfrag(b, Bs, wn * 64 + ni * 16, laneLo, hi);
#pragma unroll
            for (int mi = 0; mi < 2; ++mi) {
                acc[mi][ni] = __builtin_amdgcn_wmma_f32_16x16x32_bf16(
                    false, a[mi].v, false, b.v, (short)0, acc[mi][ni], false, false);
            }
        }
    }

#pragma unroll
    for (int mi = 0; mi < 2; ++mi)
#pragma unroll
        for (int ni = 0; ni < 4; ++ni) {
            int col = n0 + wn * 64 + ni * 16 + laneLo;
#pragma unroll
            for (int r = 0; r < 8; ++r) {
                int row = m0 + wm * 32 + mi * 16 + hi * 8 + r;
                float v = acc[mi][ni][r];
                if (rowScale) v *= rowScale[row];
                size_t o = (size_t)row * N + col;
                if (accumulate) out[o] += v; else out[o] = v;
            }
        }
}

// ---------------------------------------------------------------------------
// Launcher
// ---------------------------------------------------------------------------
extern "C" void kernel_launch(void* const* d_in, const int* in_sizes, int n_in,
                              void* d_out, int out_size, void* d_ws, size_t ws_size,
                              hipStream_t stream)
{
    (void)in_sizes; (void)n_in; (void)out_size; (void)ws_size;

    const float* x    = (const float*)d_in[0];  // [T, H]
    const float* Wr   = (const float*)d_in[1];  // [E, H]
    const float* Wsg  = (const float*)d_in[2];  // [1, H]
    const float* WsGU = (const float*)d_in[3];  // [H, 2*I_SH]
    const float* WsD  = (const float*)d_in[4];  // [I_SH, H]
    const float* WeGU = (const float*)d_in[5];  // [E, H, 2*I_MOE]
    const float* WeD  = (const float*)d_in[6];  // [E, I_MOE, H]
    float* out = (float*)d_out;                 // [T, H]

    float* ws      = (float*)d_ws;
    float* combine = ws;                                   // T*E
    float* sgate   = combine + (size_t)T_TOK * N_EXP;      // T
    float* h_sh    = sgate + T_TOK;                        // T*I_SH
    float* h_e     = h_sh + (size_t)T_TOK * I_SH;          // T*I_MOE

    dim3 blk(256);

    // 1) router probs/top-2 -> combine; shared-expert gate sigmoid
    router_kernel<<<T_TOK, 256, 0, stream>>>(x, Wr, Wsg, combine, sgate);

    // 2) shared expert: gate/up + silu*up -> h_sh ; down (*sigmoid) -> out
    gemm_gateup_silu<<<dim3(I_SH / BN, T_TOK / BM), blk, 0, stream>>>(
        x, WsGU, h_sh, nullptr, H_DIM, I_SH, 2 * I_SH, 0);
    gemm_down<<<dim3(H_DIM / BN, T_TOK / BM), blk, 0, stream>>>(
        h_sh, WsD, out, sgate, I_SH, H_DIM, /*accumulate=*/0);

    // 3) experts (dense-dispatch math): combine weight folded into h_e rows,
    //    down-proj accumulates into out. Sequential launches -> no races.
    for (int e = 0; e < N_EXP; ++e) {
        gemm_gateup_silu<<<dim3(I_MOE / BN, T_TOK / BM), blk, 0, stream>>>(
            x, WeGU + (size_t)e * H_DIM * 2 * I_MOE, h_e,
            combine + e, H_DIM, I_MOE, 2 * I_MOE, N_EXP);
        gemm_down<<<dim3(H_DIM / BN, T_TOK / BM), blk, 0, stream>>>(
            h_e, WeD + (size_t)e * I_MOE * H_DIM, out, nullptr,
            I_MOE, H_DIM, /*accumulate=*/1);
    }
}
